// HGT_9388798509139
// MI455X (gfx1250) — compile-verified, matrix operand order checked by reading, not run on hardware
//
#include <hip/hip_runtime.h>
#include <stdint.h>

typedef __attribute__((ext_vector_type(16))) _Float16 v16h;
typedef __attribute__((ext_vector_type(8)))  _Float16 v8h;
typedef __attribute__((ext_vector_type(8)))  float    v8f;
typedef __attribute__((ext_vector_type(2)))  float    v2f;

#define ACT_NONE 0
#define ACT_RELU 1
#define ACT_SKIP 2

// ---------------------------------------------------------------------------
// Load a 16x32 f16 A-fragment half-row for v_wmma_f32_16x16x32_f16 from fp32.
// Per CDNA5 ISA 7.12.2 (16-bit A-matrix 16x32): caller passes
// p = rowBase + kb*32 + half*8; elements 0..7 are K=+0..7, 8..15 are K=+16..23.
// ---------------------------------------------------------------------------
__device__ __forceinline__ v16h load_fragA(const float* __restrict__ p) {
  float4 r0 = *(const float4*)(p);
  float4 r1 = *(const float4*)(p + 4);
  float4 r2 = *(const float4*)(p + 16);
  float4 r3 = *(const float4*)(p + 20);
  v16h f;
  f[0]  = (_Float16)r0.x; f[1]  = (_Float16)r0.y; f[2]  = (_Float16)r0.z; f[3]  = (_Float16)r0.w;
  f[4]  = (_Float16)r1.x; f[5]  = (_Float16)r1.y; f[6]  = (_Float16)r1.z; f[7]  = (_Float16)r1.w;
  f[8]  = (_Float16)r2.x; f[9]  = (_Float16)r2.y; f[10] = (_Float16)r2.z; f[11] = (_Float16)r2.w;
  f[12] = (_Float16)r3.x; f[13] = (_Float16)r3.y; f[14] = (_Float16)r3.z; f[15] = (_Float16)r3.w;
  return f;
}

// B-fragment from LDS-staged f16 weights: two 16B runs (ds_load_b128 pair).
__device__ __forceinline__ v16h load_fragB_lds(const _Float16* wp) {
  v8h b0 = *(const v8h*)(wp);
  v8h b1 = *(const v8h*)(wp + 16);
  v16h f;
#pragma unroll
  for (int j = 0; j < 8; ++j) { f[j] = b0[j]; f[8 + j] = b1[j]; }
  return f;
}

// ---------------------------------------------------------------------------
// Y[M,NOUT] = act(X[M,128] @ W[NOUT,128]^T + B), one 16-row tile per wave,
// K=128 -> 4x v_wmma_f32_16x16x32_f16, fp32 accumulate.  W is staged once per
// workgroup into LDS as f16 with 136-elem row stride (272B -> bank-conflict-
// free b128 reads).  MODE==ACT_SKIP: Y = s*(XW^T+B) + (1-s)*Xold, s=sigmoid.
// ---------------------------------------------------------------------------
template <int NOUT, int MODE>
__global__ void __launch_bounds__(128)
gemm_wmma(const float* __restrict__ X, const float* __restrict__ W,
          const float* __restrict__ B, const float* __restrict__ Xold,
          const float* __restrict__ Skip, float* __restrict__ Y,
          int Mtiles, int Nrows) {
  constexpr int LDW = 136;               // 128 + 8 f16 pad
  __shared__ _Float16 sW[NOUT * LDW];
  const int tid = threadIdx.x;
  for (int i = tid; i < NOUT * 32; i += 128) {   // cooperative W -> LDS (f16)
    const int n  = i >> 5;
    const int k4 = (i & 31) << 2;
    float4 wv = *(const float4*)(W + (size_t)n * 128 + k4);
    _Float16* d = sW + n * LDW + k4;
    d[0] = (_Float16)wv.x; d[1] = (_Float16)wv.y;
    d[2] = (_Float16)wv.z; d[3] = (_Float16)wv.w;
  }
  __syncthreads();                        // all waves reach this before exit

  const int mtile = blockIdx.x * 4 + (tid >> 5);
  if (mtile >= Mtiles) return;
  const int lane = tid & 31;
  const int half = lane >> 4;
  const int mr   = lane & 15;
  constexpr int NT = NOUT / 16;
  v8f acc[NT] = {};
  int row = mtile * 16 + mr;
  if (row >= Nrows) row = Nrows - 1;
#pragma unroll
  for (int kb = 0; kb < 4; ++kb) {
    const int koff = kb * 32 + half * 8;
    v16h a = load_fragA(X + (size_t)row * 128 + koff);
#pragma unroll
    for (int nt = 0; nt < NT; ++nt) {
      v16h b = load_fragB_lds(sW + (nt * 16 + mr) * LDW + koff);
      acc[nt] = __builtin_amdgcn_wmma_f32_16x16x32_f16(
          false, a, false, b, (short)0, acc[nt], false, false);
    }
  }
  float al = 0.0f;
  if (MODE == ACT_SKIP) al = 1.0f / (1.0f + expf(-Skip[0]));
  const bool full = (mtile * 16 + 16) <= Nrows;
#pragma unroll
  for (int nt = 0; nt < NT; ++nt) {
    const int col = nt * 16 + mr;
    const float bb = B[col];
    if (full) {
#pragma unroll
      for (int r = 0; r < 8; ++r) {
        const int orow = mtile * 16 + half * 8 + r;   // C/D layout: M = r+8*half
        float v = acc[nt][r] + bb;
        if (MODE == ACT_RELU) v = v > 0.0f ? v : 0.0f;
        if (MODE == ACT_SKIP)
          v = al * v + (1.0f - al) * Xold[(size_t)orow * NOUT + col];
        Y[(size_t)orow * NOUT + col] = v;
      }
    } else {
#pragma unroll
      for (int r = 0; r < 8; ++r) {
        const int orow = mtile * 16 + half * 8 + r;
        if (orow < Nrows) {
          float v = acc[nt][r] + bb;
          if (MODE == ACT_RELU) v = v > 0.0f ? v : 0.0f;
          if (MODE == ACT_SKIP)
            v = al * v + (1.0f - al) * Xold[(size_t)orow * NOUT + col];
          Y[(size_t)orow * NOUT + col] = v;
        }
      }
    }
  }
}

// ---------------------------------------------------------------------------
// Per-head relation projection: KR[n,h,:] = K[n,h,:] @ Arel[h]  (16x16, fp32),
// VR likewise with Mrel. One (16-node-tile, head) pair per wave.
// K=16 -> 4x v_wmma_f32_16x16x4_f32 (A 16x4 layout: lanes0-15 K=0,1;
// lanes16-31 K=2,3 per CDNA5 ISA 7.12.2).
// ---------------------------------------------------------------------------
__global__ void __launch_bounds__(128)
relproj_wmma(const float* __restrict__ K, const float* __restrict__ V,
             const float* __restrict__ Arel, const float* __restrict__ Mrel,
             float* __restrict__ KR, float* __restrict__ VR,
             int Mtiles, int Nrows) {
  const int pair = blockIdx.x * (blockDim.x >> 5) + (threadIdx.x >> 5);
  if (pair >= Mtiles * 8) return;
  const int mtile = pair >> 3;
  const int h     = pair & 7;
  const int lane  = threadIdx.x & 31;
  const int half  = lane >> 4;
  const int mr    = lane & 15;
  int row = mtile * 16 + mr;
  if (row >= Nrows) row = Nrows - 1;
  const float* krow = K + (size_t)row * 128 + h * 16;
  const float* vrow = V + (size_t)row * 128 + h * 16;
  v8f acck = {}, accv = {};
#pragma unroll
  for (int kb = 0; kb < 4; ++kb) {
    const int k0 = kb * 4 + half * 2;
    v2f ak, av, bk, bv;
    ak.x = krow[k0];  ak.y = krow[k0 + 1];
    av.x = vrow[k0];  av.y = vrow[k0 + 1];
    bk.x = Arel[(h * 16 + k0) * 16 + mr];
    bk.y = Arel[(h * 16 + k0 + 1) * 16 + mr];
    bv.x = Mrel[(h * 16 + k0) * 16 + mr];
    bv.y = Mrel[(h * 16 + k0 + 1) * 16 + mr];
    acck = __builtin_amdgcn_wmma_f32_16x16x4_f32(false, ak, false, bk, (short)0, acck, false, false);
    accv = __builtin_amdgcn_wmma_f32_16x16x4_f32(false, av, false, bv, (short)0, accv, false, false);
  }
  const bool full = (mtile * 16 + 16) <= Nrows;
  if (full) {
#pragma unroll
    for (int r = 0; r < 8; ++r) {
      const int orow = mtile * 16 + half * 8 + r;
      KR[(size_t)orow * 128 + h * 16 + mr] = acck[r];
      VR[(size_t)orow * 128 + h * 16 + mr] = accv[r];
    }
  } else {
#pragma unroll
    for (int r = 0; r < 8; ++r) {
      const int orow = mtile * 16 + half * 8 + r;
      if (orow < Nrows) {
        KR[(size_t)orow * 128 + h * 16 + mr] = acck[r];
        VR[(size_t)orow * 128 + h * 16 + mr] = accv[r];
      }
    }
  }
}

// ---------------------------------------------------------------------------
// Edge passes
// ---------------------------------------------------------------------------
__device__ __forceinline__ void atomicMaxFloat(float* addr, float val) {
  // IEEE bit-order trick: int max for >=0, uint min for <0 (init = -inf bits).
  if (val >= 0.0f) atomicMax((int*)addr, __float_as_int(val));
  else             atomicMin((unsigned int*)addr, __float_as_uint(val));
}

__global__ void fill_u32(uint32_t* __restrict__ p, uint32_t v, int n) {
  int i = blockIdx.x * blockDim.x + threadIdx.x;
  if (i < n) p[i] = v;
}

__global__ void attn_kernel(const float* __restrict__ Q, const float* __restrict__ KR,
                            const float* __restrict__ VR, const int* __restrict__ ei,
                            const float* __restrict__ prel, float* __restrict__ alpha,
                            float* __restrict__ segmax, int E) {
  int idx = blockIdx.x * blockDim.x + threadIdx.x;
  if (idx >= E * 8) return;
  const int e = idx >> 3, h = idx & 7;
  const int src = ei[e], dst = ei[E + e];
  const float4* qp = (const float4*)(Q  + (size_t)dst * 128 + h * 16);
  const float4* kp = (const float4*)(KR + (size_t)src * 128 + h * 16);
  __builtin_prefetch(VR + (size_t)src * 128 + h * 16, 0, 0);  // warm L2 for aggregate pass
  float s = 0.0f;
#pragma unroll
  for (int i = 0; i < 4; ++i) {
    float4 q4 = qp[i], k4 = kp[i];
    s += q4.x * k4.x + q4.y * k4.y + q4.z * k4.z + q4.w * k4.w;
  }
  s *= prel[h] * 0.25f;  // SCALE = 1/sqrt(16)
  alpha[idx] = s;
  atomicMaxFloat(segmax + (size_t)dst * 8 + h, s);
}

__global__ void expsum_kernel(const int* __restrict__ ei, float* __restrict__ alpha,
                              const float* __restrict__ segmax, float* __restrict__ segsum,
                              int E) {
  int idx = blockIdx.x * blockDim.x + threadIdx.x;
  if (idx >= E * 8) return;
  const int e = idx >> 3, h = idx & 7;
  const int dst = ei[E + e];
  float m = segmax[(size_t)dst * 8 + h];
  if (!(m >= -3.0e38f && m <= 3.0e38f)) m = 0.0f;  // isfinite clamp as in reference
  float ex = expf(alpha[idx] - m);
  alpha[idx] = ex;
  atomicAdd(segsum + (size_t)dst * 8 + h, ex);
}

__global__ void aggregate_kernel(const int* __restrict__ ei, const float* __restrict__ alpha,
                                 const float* __restrict__ segsum, const float* __restrict__ VR,
                                 float* __restrict__ agg, int E) {
  int idx = blockIdx.x * blockDim.x + threadIdx.x;
  if (idx >= E * 128) return;
  const int e = idx >> 7, c = idx & 127, h = c >> 4;
  const int src = ei[e], dst = ei[E + e];
  const float w = alpha[(size_t)e * 8 + h] / (segsum[(size_t)dst * 8 + h] + 1e-16f);
  atomicAdd(agg + (size_t)dst * 128 + c, VR[(size_t)src * 128 + c] * w);
}

__global__ void gelu_kernel(const float* __restrict__ in, float* __restrict__ out, int n) {
  int i = blockIdx.x * blockDim.x + threadIdx.x;
  if (i < n) {
    float x = in[i];
    out[i] = 0.5f * x * (1.0f + erff(x * 0.70710678118654752f));  // exact GELU
  }
}

// ---------------------------------------------------------------------------
// Launcher. Input flattening follows setup_inputs() dict insertion order:
//  0:x_a 1:x_b 2,3:in_w(a,b) 4,5:in_b 6,7:out_w 8,9:out_b
//  10+27*l: conv l = [k_w a,b, k_b a,b, q_w a,b, q_b a,b, v_w a,b, v_b a,b,
//                     a_w a,b, a_b a,b, skip a,b, a_rel x3, m_rel x3, p_rel x3]
//  64,65,66: ei_ab, ei_ba, ei_bb
// ---------------------------------------------------------------------------
extern "C" void kernel_launch(void* const* d_in, const int* in_sizes, int n_in,
                              void* d_out, int out_size, void* d_ws, size_t ws_size,
                              hipStream_t stream) {
  (void)n_in; (void)out_size; (void)ws_size;
  auto P = [&](int i) { return (const float*)d_in[i]; };
  const int NA = in_sizes[0] / 128;
  const int NB = in_sizes[1] / 128;
  const int TA = (NA + 15) / 16;
  const int TB = (NB + 15) / 16;
  const int CONV = 10, CS = 27, EIB = CONV + 2 * CS;  // 64
  const int* eis[3] = {(const int*)d_in[EIB + 0], (const int*)d_in[EIB + 1],
                       (const int*)d_in[EIB + 2]};
  int Es[3] = {in_sizes[EIB + 0] / 2, in_sizes[EIB + 1] / 2, in_sizes[EIB + 2] / 2};
  int maxE = Es[0] > Es[1] ? Es[0] : Es[1]; if (Es[2] > maxE) maxE = Es[2];
  const size_t maxN = (size_t)(NA > NB ? NA : NB);

  float* w = (float*)d_ws;
  auto take = [&](size_t n) { float* p = w; w += n; return p; };
  float* xa   = take((size_t)NA * 128);
  float* xb   = take((size_t)NB * 128);
  float* ka   = take((size_t)NA * 128);
  float* kb   = take((size_t)NB * 128);
  float* qa   = take((size_t)NA * 128);
  float* qb   = take((size_t)NB * 128);
  float* va   = take((size_t)NA * 128);
  float* vb   = take((size_t)NB * 128);
  float* kr   = take(maxN * 128);
  float* vr   = take(maxN * 128);
  float* alp  = take((size_t)maxE * 8);
  float* smax = take(maxN * 8);
  float* ssum = take(maxN * 8);
  float* agga = take((size_t)NA * 128);
  float* aggb = take((size_t)NB * 128);

  const dim3 blk(128);
  auto g4 = [](int tiles) { return dim3((unsigned)((tiles + 3) / 4)); };
  auto fill = [&](float* p, uint32_t bits, int n) {
    fill_u32<<<dim3((unsigned)((n + 255) / 256)), dim3(256), 0, stream>>>((uint32_t*)p, bits, n);
  };

  // input linears: x_t = relu(x_in @ in_w^T + in_b)
  gemm_wmma<128, ACT_RELU><<<g4(TA), blk, 0, stream>>>(P(0), P(2), P(4), nullptr, nullptr, xa, TA, NA);
  gemm_wmma<128, ACT_RELU><<<g4(TB), blk, 0, stream>>>(P(1), P(3), P(5), nullptr, nullptr, xb, TB, NB);

  for (int l = 0; l < 2; ++l) {
    const int cb = CONV + l * CS;
    // k/q/v for both node types
    gemm_wmma<128, ACT_NONE><<<g4(TA), blk, 0, stream>>>(xa, P(cb + 0), P(cb + 2),  nullptr, nullptr, ka, TA, NA);
    gemm_wmma<128, ACT_NONE><<<g4(TB), blk, 0, stream>>>(xb, P(cb + 1), P(cb + 3),  nullptr, nullptr, kb, TB, NB);
    gemm_wmma<128, ACT_NONE><<<g4(TA), blk, 0, stream>>>(xa, P(cb + 4), P(cb + 6),  nullptr, nullptr, qa, TA, NA);
    gemm_wmma<128, ACT_NONE><<<g4(TB), blk, 0, stream>>>(xb, P(cb + 5), P(cb + 7),  nullptr, nullptr, qb, TB, NB);
    gemm_wmma<128, ACT_NONE><<<g4(TA), blk, 0, stream>>>(xa, P(cb + 8), P(cb + 10), nullptr, nullptr, va, TA, NA);
    gemm_wmma<128, ACT_NONE><<<g4(TB), blk, 0, stream>>>(xb, P(cb + 9), P(cb + 11), nullptr, nullptr, vb, TB, NB);

    fill(agga, 0u, NA * 128);
    fill(aggb, 0u, NB * 128);

    // edge types: (a->b), (b->a), (b->b)
    const float* ksrc[3] = {ka, kb, kb};
    const float* vsrc[3] = {va, vb, vb};
    const float* qdst[3] = {qb, qa, qb};
    float*       aggd[3] = {aggb, agga, aggb};
    int stile[3] = {TA, TB, TB};
    int snod[3]  = {NA, NB, NB};
    int dnod[3]  = {NB, NA, NB};
    for (int t = 0; t < 3; ++t) {
      relproj_wmma<<<g4(stile[t] * 8), blk, 0, stream>>>(
          ksrc[t], vsrc[t], P(cb + 18 + t), P(cb + 21 + t), kr, vr, stile[t], snod[t]);
      fill(smax, 0xFF800000u, dnod[t] * 8);  // -inf
      fill(ssum, 0u, dnod[t] * 8);
      const int EH = Es[t] * 8;
      attn_kernel<<<dim3((unsigned)((EH + 255) / 256)), dim3(256), 0, stream>>>(
          qdst[t], kr, vr, eis[t], P(cb + 24 + t), alp, smax, Es[t]);
      expsum_kernel<<<dim3((unsigned)((EH + 255) / 256)), dim3(256), 0, stream>>>(
          eis[t], alp, smax, ssum, Es[t]);
      const int EC = Es[t] * 128;
      aggregate_kernel<<<dim3((unsigned)((EC + 255) / 256)), dim3(256), 0, stream>>>(
          eis[t], alp, ssum, vr, aggd[t], Es[t]);
    }

    // new x = sigmoid(skip)*(gelu(agg) @ a_w^T + a_b) + (1-sigmoid(skip))*x
    gelu_kernel<<<dim3((unsigned)((NA * 128 + 255) / 256)), dim3(256), 0, stream>>>(agga, ka, NA * 128);
    gelu_kernel<<<dim3((unsigned)((NB * 128 + 255) / 256)), dim3(256), 0, stream>>>(aggb, kb, NB * 128);
    gemm_wmma<128, ACT_SKIP><<<g4(TA), blk, 0, stream>>>(ka, P(cb + 12), P(cb + 14), xa, P(cb + 16), xa, TA, NA);
    gemm_wmma<128, ACT_SKIP><<<g4(TB), blk, 0, stream>>>(kb, P(cb + 13), P(cb + 15), xb, P(cb + 17), xb, TB, NB);
  }

  // output heads: relu(x_t @ out_w^T + out_b) -> d_out (a then b, flat)
  float* out = (float*)d_out;
  gemm_wmma<64, ACT_RELU><<<g4(TA), blk, 0, stream>>>(xa, P(6), P(8), nullptr, nullptr, out, TA, NA);
  gemm_wmma<64, ACT_RELU><<<g4(TB), blk, 0, stream>>>(xb, P(7), P(9), nullptr, nullptr, out + (size_t)NA * 64, TB, NB);
}
